// DHT_45002667327997
// MI455X (gfx1250) — matrix operand order repaired and checked by reading, not compiled, not run on hardware
//
#include <hip/hip_runtime.h>
#include <hip/hip_bf16.h>
#include <math.h>
#include <type_traits>

// ---------------------------------------------------------------------------
// Deep Hough Transform on gfx1250 (MI455X).
//   feat: [N=4, C=128, H=256, W=256] f32  ->  out: [4,128,180,723] f32
//
// Scatter-add recast as one-hot WMMA: for a 32-pixel run along x at fixed
// angle, rho bins span <= 32, so
//   D[16 rho x 16 ch] += Aonehot[16 x 32] * B[32 px x 16 ch]
// with v_wmma_f32_16x16x32_f16. The rho window is chained in the WMMA
// accumulators across chunks and flushed to an LDS histogram (ds_add_f32)
// only when it drifts out of range.
//
// v4: window state (rbase / rebase / need1) is forced into SGPRs via
// readfirstlane so WMMAs are never under a masked EXEC (ISA: EXEC must be
// all-1s for WMMA) and the hot loop uses scalar branches instead of
// s_and_saveexec. feat is converted to f16 once (ws-guarded pre-pass); each
// block runs NT=2 channel tiles off one A build; LDS histogram 756x33 floats.
// ---------------------------------------------------------------------------

typedef __attribute__((ext_vector_type(16))) _Float16 v16h;
typedef __attribute__((ext_vector_type(8)))  _Float16 v8h;
typedef __attribute__((ext_vector_type(8)))  float    v8f;

#define DHT_H      256
#define DHT_W      256
#define DHT_P      65536        // H*W
#define DHT_NC     512          // N*C
#define DHT_NA     180
#define DHT_NR     723
#define DHT_RPAD   756          // 723 + pad so window-1 rows never go OOB
#define DHT_NT     2            // channel tiles (of 16) per block
#define DHT_CSTR   (DHT_NT * 16 + 1)  // LDS row stride (odd -> conflict-free)
#define DHT_RHALF  361          // numrho/2
#define DHT_PI_F   3.14159265358979323846f

// ---- Pre-pass: feat f32 -> f16 (one-time; removes per-angle reconversion) --
__global__ __launch_bounds__(256) void
DHT_cvt_f16_kernel(const float* __restrict__ in, _Float16* __restrict__ out)
{
    const int i = (blockIdx.x * 256 + threadIdx.x) * 8;
    const float4 f0 = *(const float4*)(in + i);
    const float4 f1 = *(const float4*)(in + i + 4);
    v8h h;
    h[0] = (_Float16)f0.x; h[1] = (_Float16)f0.y;
    h[2] = (_Float16)f0.z; h[3] = (_Float16)f0.w;
    h[4] = (_Float16)f1.x; h[5] = (_Float16)f1.y;
    h[6] = (_Float16)f1.z; h[7] = (_Float16)f1.w;
    *(v8h*)(out + i) = h;
}

// ---- Main kernel (T = _Float16 fast path, or float fallback) ---------------
template <typename T>
__global__ __launch_bounds__(256) void
DHT_45002667327997_kernel(const T* __restrict__ feat, float* __restrict__ out)
{
    __shared__ float acc[DHT_RPAD * DHT_CSTR];   // ~99.8 KB of 320 KB WGP LDS

    const int ctg = blockIdx.x;          // channel-tile group: NT tiles of 16
    const int ang = blockIdx.y;          // angle 0..179
    const int tid = threadIdx.x;

    for (int i = tid; i < DHT_RPAD * DHT_CSTR; i += 256) acc[i] = 0.0f;
    __syncthreads();

    const float th  = (float)ang * (DHT_PI_F / (float)DHT_NA);
    const float cth = __cosf(th);
    const float sth = __sinf(th);
    const float inv = 1.0f / cth;        // finite even near 90 deg (cth ~ -7e-8)

    const int lane = tid & 31;
    const int wave = tid >> 5;
    const int m    = lane & 15;          // A row / B,D column owned by this lane
    const int hi   = lane >> 4;          // 0: low half of K, 1: high half

    const T* frow[DHT_NT];
#pragma unroll
    for (int t = 0; t < DHT_NT; ++t)
        frow[t] = feat + (size_t)((ctg * DHT_NT + t) * 16 + m) * DHT_P;

    v8f  c0[DHT_NT];                     // window [rbase, rbase+16)
    v8f  c1[DHT_NT];                     // window [rbase+16, rbase+32)
#pragma unroll
    for (int t = 0; t < DHT_NT; ++t) { c0[t] = (v8f)0.0f; c1[t] = (v8f)0.0f; }
    int  rbase = -0x40000000;            // kept wave-uniform (readfirstlane)
    int  used1 = 0;                      // scalar: window-1 touched since flush

    auto flushWin = [&]() {
#pragma unroll
        for (int t = 0; t < DHT_NT; ++t) {
#pragma unroll
            for (int v = 0; v < 8; ++v)
                atomicAdd(&acc[(rbase + v + hi * 8) * DHT_CSTR + t * 16 + m],
                          c0[t][v]);
            c0[t] = (v8f)0.0f;
        }
        if (used1) {
#pragma unroll
            for (int t = 0; t < DHT_NT; ++t) {
#pragma unroll
                for (int v = 0; v < 8; ++v)
                    atomicAdd(&acc[(rbase + 16 + v + hi * 8) * DHT_CSTR + t * 16 + m],
                              c1[t][v]);
                c1[t] = (v8f)0.0f;
            }
        }
        used1 = 0;
    };

    // 2048 chunks of 32 pixels; each wave owns a contiguous range so the rho
    // window drifts slowly and WMMA keeps accumulating in registers.
    const int cBeg = wave * 256;
    const int cEnd = cBeg + 256;

    for (int chk = cBeg; chk < cEnd; ++chk) {
        const int   pb   = chk << 5;                       // one y row per chunk
        const int   y    = pb >> 8;
        const float rho0 = (float)((pb & 255) - (DHT_W / 2)) * cth
                         + (float)(y - (DHT_H / 2)) * sth
                         + (float)DHT_RHALF;               // bin space

        const int rA   = min(DHT_NR - 1, max(0, (int)rintf(rho0)));
        const int rB   = min(DHT_NR - 1, max(0, (int)rintf(rho0 + 31.0f * cth)));
        // All lanes compute identical values; pin them to SGPRs so the window
        // control flow is scalar branches (EXEC untouched around WMMA).
        const int rmin = __builtin_amdgcn_readfirstlane(min(rA, rB));
        const int rmax = __builtin_amdgcn_readfirstlane(max(rA, rB));

        if (rmin < rbase || rmax >= rbase + 32) {          // scalar branch
            if (rbase >= 0) flushWin();
            rbase = rmin;
        }

        // ---- Interval one-hot masks: bit K of maskX set iff pixel K maps to
        // this lane's row. Half-open [ceil(lo), ceil(hi)) avoids overlap.
        auto rowMask = [&](int rv) -> unsigned {
            if (rv > DHT_NR - 1) return 0u;                // rows past clip: empty
            const float lowB = (rv == 0)          ? -1.0e30f : ((float)rv - 0.5f);
            const float hiB  = (rv == DHT_NR - 1) ?  1.0e30f : ((float)rv + 0.5f);
            const float t0 = (lowB - rho0) * inv;
            const float t1 = (hiB  - rho0) * inv;
            const float lo = fmaxf(fminf(t0, t1), -1.0f);
            const float hc = fminf(fmaxf(t0, t1), 33.0f);
            const int kLo = max(0,  (int)ceilf(lo));
            const int kHi = min(32, (int)ceilf(hc));
            if (kHi <= kLo) return 0u;
            return (unsigned)(((1ull << kHi) - 1ull) & ~((1ull << kLo) - 1ull));
        };
        const unsigned mask0 = rowMask(rbase + m);
        const unsigned mask1 = rowMask(rbase + 16 + m);

        // ---- A tiles from the row masks (this code's contiguous K order):
        // lanes 0-15: e=0..15 -> K=e; lanes 16-31: K=16+e (matches B).
        v16h a0, a1;
#pragma unroll
        for (int e = 0; e < 16; ++e) {
            const int K = e + hi * 16;
            a0[e] = ((mask0 >> K) & 1u) ? (_Float16)1.0f : (_Float16)0.0f;
            a1[e] = ((mask1 >> K) & 1u) ? (_Float16)1.0f : (_Float16)0.0f;
        }
        const int need1 = (rmax - rbase) >= 16;            // scalar (rmin/rmax
        used1 |= need1;                                    //  are SGPRs)

        // ---- One A, NT B tiles / WMMA streams (amortizes the A build).
        // B in the same contiguous K order as A: lane(N=m) holds K = hi*16 + 0..15.
#pragma unroll
        for (int t = 0; t < DHT_NT; ++t) {
            v16h b;
            if constexpr (std::is_same<T, _Float16>::value) {
                b = *(const v16h*)(frow[t] + pb + hi * 16); // 32 B: 2x b128
            } else {
                const float4* src = (const float4*)(frow[t] + pb + hi * 16);
#pragma unroll
                for (int q = 0; q < 4; ++q) {
                    float4 f = src[q];
                    b[q * 4 + 0] = (_Float16)f.x;
                    b[q * 4 + 1] = (_Float16)f.y;
                    b[q * 4 + 2] = (_Float16)f.z;
                    b[q * 4 + 3] = (_Float16)f.w;
                }
            }
            if (t == 0)
                __builtin_prefetch(frow[0] + pb + 32, 0, 3);   // next chunk

            c0[t] = __builtin_amdgcn_wmma_f32_16x16x32_f16(false, a0, false, b,
                                                           (short)0, c0[t],
                                                           false, false);
            if (need1)                                     // scalar branch
                c1[t] = __builtin_amdgcn_wmma_f32_16x16x32_f16(false, a1, false, b,
                                                               (short)0, c1[t],
                                                               false, false);
        }
    }
    if (rbase >= 0) flushWin();

    __syncthreads();

    // ---- Write back LDS histogram -> out[ch][ang][rho], coalesced over rho,
    // LDS reads conflict-free thanks to the odd (33-float) row stride.
    const int abase = ang * DHT_NR;
#pragma unroll 1
    for (int j = 0; j < DHT_NT * 16; ++j) {
        float* orow = out + (size_t)(ctg * DHT_NT * 16 + j) * (DHT_NA * DHT_NR)
                    + abase;
        for (int r = tid; r < DHT_NR; r += 256)
            orow[r] = acc[r * DHT_CSTR + j];
    }
}

extern "C" void kernel_launch(void* const* d_in, const int* in_sizes, int n_in,
                              void* d_out, int out_size, void* d_ws, size_t ws_size,
                              hipStream_t stream)
{
    (void)in_sizes; (void)n_in; (void)out_size;
    const float* feat = (const float*)d_in[0];
    float*       out  = (float*)d_out;

    dim3 grid(32 / DHT_NT, DHT_NA);   // (channel-tile groups) x (angles)
    dim3 block(256);                  // 8 waves of 32

    const size_t f16Bytes = (size_t)DHT_NC * DHT_P * sizeof(_Float16); // 64 MB
    if (ws_size >= f16Bytes) {
        _Float16* feat16 = (_Float16*)d_ws;
        const int nElem  = DHT_NC * DHT_P;
        DHT_cvt_f16_kernel<<<nElem / (256 * 8), 256, 0, stream>>>(feat, feat16);
        DHT_45002667327997_kernel<_Float16><<<grid, block, 0, stream>>>(feat16, out);
    } else {
        DHT_45002667327997_kernel<float><<<grid, block, 0, stream>>>(feat, out);
    }
}
